// FAST_28467043237937
// MI455X (gfx1250) — matrix operand order repaired and checked by reading, not compiled, not run on hardware
//
#include <hip/hip_runtime.h>

// ---------------------------------------------------------------------------
// Problem constants
// ---------------------------------------------------------------------------
#define S_  88
#define L_  10
#define T_  30
#define E_  768
#define H_  64
#define G4_ 256          // 4*H
#define M_  300          // L*T rows per stock in the big GEMM

typedef __attribute__((ext_vector_type(16))) _Float16 v16h;
typedef __attribute__((ext_vector_type(8)))  _Float16 v8h;
typedef __attribute__((ext_vector_type(8)))  float    v8f;

__device__ __forceinline__ v8f vzero8() {
    v8f z;
#pragma unroll
    for (int i = 0; i < 8; ++i) z[i] = 0.0f;
    return z;
}

__device__ __forceinline__ float sigmoidf_(float x) {
    return 1.0f / (1.0f + __expf(-x));
}

__device__ __forceinline__ v16h combine16(v8h lo, v8h hh) {
    v16h out;
#pragma unroll
    for (int i = 0; i < 8; ++i) { out[i] = lo[i]; out[8 + i] = hh[i]; }
    return out;
}

// ---------------------------------------------------------------------------
// WMMA fragment loads (f16, 16x16x32 shape), from any address space.
//
// A operand, 16x32, row-major tile [rows][ld]:
//   lane l<16 : row=l,    K = kbase+0..7  and kbase+16..23
//   lane l>=16: row=l-16, K = kbase+8..15 and kbase+24..31
// (ISA 7.12.2 "16-bit A-Matrix 16x32")
//
// B operand, 32x16, stored transposed as Bt[N][K]:
//   lane l<16 : N=l,    K = kbase+0..7  and kbase+16..23
//   lane l>=16: N=l-16, K = kbase+8..15 and kbase+24..31
// (same K order as the A operand)
// ---------------------------------------------------------------------------
__device__ __forceinline__ v16h frag_A(const _Float16* base, int ld,
                                       int row0, int kbase, int lane) {
    int r  = lane & 15;
    int hi = lane >> 4;
    const _Float16* p = base + (row0 + r) * ld + kbase + hi * 8;
    return combine16(*(const v8h*)p, *(const v8h*)(p + 16));
}

__device__ __forceinline__ v16h frag_B(const _Float16* baseT, int ld,
                                       int col0, int kbase, int lane) {
    int n  = lane & 15;
    int hi = lane >> 4;
    const _Float16* p = baseT + (col0 + n) * ld + kbase + hi * 8;
    return combine16(*(const v8h*)p, *(const v8h*)(p + 16));
}

// ===========================================================================
// Kernel 0a: text_inputs fp32 -> f16 (same layout). 8 elems / thread.
// ===========================================================================
__global__ __launch_bounds__(256) void conv_x(const float* __restrict__ X,
                                              _Float16* __restrict__ Xh) {
    size_t base = ((size_t)blockIdx.x * 256 + threadIdx.x) * 8;
    float4 a = *(const float4*)(X + base);
    float4 b = *(const float4*)(X + base + 4);
    v8h o;
    o[0] = (_Float16)a.x; o[1] = (_Float16)a.y;
    o[2] = (_Float16)a.z; o[3] = (_Float16)a.w;
    o[4] = (_Float16)b.x; o[5] = (_Float16)b.y;
    o[6] = (_Float16)b.z; o[7] = (_Float16)b.w;
    *(v8h*)(Xh + base) = o;
}

// ===========================================================================
// Kernel 0b: Uall_w [S][768][256] fp32 -> Wt [S][256][768] f16 (transposed).
// 64x64 tile per block via LDS. grid (48, 88): tile = x (12 k-tiles x 4 n-tiles)
// ===========================================================================
__global__ __launch_bounds__(256) void conv_w(const float* __restrict__ W,
                                              _Float16* __restrict__ Wt) {
    __shared__ _Float16 Tl[64 * 72];
    const int s  = blockIdx.y;
    const int kt = blockIdx.x % 12, nt = blockIdx.x / 12;
    const int k0 = kt * 64, n0 = nt * 64;
    const int tid = threadIdx.x;
    const float*    Ws  = W  + (size_t)s * E_ * G4_;
    _Float16*       WtS = Wt + (size_t)s * G4_ * E_;
#pragma unroll
    for (int it = 0; it < 4; ++it) {            // coalesced fp32 read, f16 LDS T
        int j = tid + it * 256;                 // 1024 float4
        int kloc = j >> 4;
        int nloc = (j & 15) * 4;
        float4 v = *(const float4*)(Ws + (size_t)(k0 + kloc) * G4_ + n0 + nloc);
        Tl[(nloc + 0) * 72 + kloc] = (_Float16)v.x;
        Tl[(nloc + 1) * 72 + kloc] = (_Float16)v.y;
        Tl[(nloc + 2) * 72 + kloc] = (_Float16)v.z;
        Tl[(nloc + 3) * 72 + kloc] = (_Float16)v.w;
    }
    __syncthreads();
#pragma unroll
    for (int it = 0; it < 2; ++it) {            // contiguous f16 writes (b128)
        int j = tid + it * 256;                 // 512 v8h
        int nl  = j >> 3;
        int kl8 = (j & 7) * 8;
        *(v8h*)(WtS + (size_t)(n0 + nl) * E_ + k0 + kl8) =
            *(const v8h*)&Tl[nl * 72 + kl8];
    }
}

// ===========================================================================
// Kernel 1: u[s] = X[s] (300x768) @ Uall_w[s] (768x256) + Uall_b[s]
// Pure register-resident WMMA GEMM streamed from L2-resident f16 operands.
// Per-wave tiling: 2 row-tiles x 4 col-tiles -> every B frag feeds 2 WMMAs,
// every A frag feeds 4. 12 b128 loads per 8 WMMAs per 32-K panel.
// grid (5, 88), 256 threads, no LDS, no barriers.
// ===========================================================================
__global__ __launch_bounds__(256) void fast_ugemm(
    const _Float16* __restrict__ Xh, const _Float16* __restrict__ Wt,
    const float* __restrict__ B, _Float16* __restrict__ U) {
    const int s  = blockIdx.y;
    const int m0 = blockIdx.x * 64;
    const int tid  = threadIdx.x;
    const int lane = tid & 31;
    const int w    = tid >> 5;
    const int rh   = (w & 1) * 32;   // row half of the 64-row slab (2 tiles)
    const int cg   = (w >> 1) * 4;   // col-tile group base (4 tiles)
    const int r    = lane & 15;
    const int hi   = lane >> 4;

    int arow0 = m0 + rh + r;      if (arow0 > M_ - 1) arow0 = M_ - 1;
    int arow1 = m0 + rh + 16 + r; if (arow1 > M_ - 1) arow1 = M_ - 1;
    const _Float16* ap0 = Xh + ((size_t)s * M_ + arow0) * E_ + hi * 8;
    const _Float16* ap1 = Xh + ((size_t)s * M_ + arow1) * E_ + hi * 8;
    const _Float16* bw  = Wt + ((size_t)s * G4_ + (cg * 16 + r)) * E_ + hi * 8;

    v8f acc[8];                       // [ct] for row-tile0, [4+ct] for row-tile1
#pragma unroll
    for (int i = 0; i < 8; ++i) acc[i] = vzero8();

    for (int kk = 0; kk < E_ / 32; ++kk) {
        __builtin_prefetch(ap0 + 64, 0, 1);
        __builtin_prefetch(ap1 + 64, 0, 1);
        v16h a0 = combine16(*(const v8h*)ap0, *(const v8h*)(ap0 + 16));
        v16h a1 = combine16(*(const v8h*)ap1, *(const v8h*)(ap1 + 16));
#pragma unroll
        for (int ct = 0; ct < 4; ++ct) {
            const _Float16* bp = bw + (size_t)ct * 16 * E_;
            v16h b = combine16(*(const v8h*)bp, *(const v8h*)(bp + 16));
            acc[ct]     = __builtin_amdgcn_wmma_f32_16x16x32_f16(
                false, a0, false, b, (short)0, acc[ct],     false, false);
            acc[4 + ct] = __builtin_amdgcn_wmma_f32_16x16x32_f16(
                false, a1, false, b, (short)0, acc[4 + ct], false, false);
        }
        ap0 += 32; ap1 += 32; bw += 32;
    }

    _Float16* Us = U + (size_t)s * M_ * G4_;
#pragma unroll
    for (int ct = 0; ct < 4; ++ct) {
        int   n    = (cg + ct) * 16 + r;
        float bias = B[(size_t)s * G4_ + n];
#pragma unroll
        for (int v = 0; v < 8; ++v) {
            int m = m0 + rh + v + hi * 8;
            if (m < M_)      Us[(size_t)m * G4_ + n]        = (_Float16)(acc[ct][v] + bias);
            if (m + 16 < M_) Us[(size_t)(m + 16) * G4_ + n] = (_Float16)(acc[4 + ct][v] + bias);
        }
    }
}

// ===========================================================================
// Kernel 2: per-stock TimeLSTM scan + attentions + day LSTM + head.
// One block (256 thr, 8 waves) per stock; weights live in LDS as f16.
// ===========================================================================
#define LD2 72   // 64 + 8 halves pad (144 B rows)

__global__ __launch_bounds__(256) void fast_stock(
    const _Float16* __restrict__ u,     // [S,300,256] f16 from kernel 1
    const float* __restrict__ time_in,  // [S,L,T]
    const float* __restrict__ Wd_w,  const float* __restrict__ Wd_b,
    const float* __restrict__ Wall_w,const float* __restrict__ Wall_b,
    const float* __restrict__ tW1_w, const float* __restrict__ tW1_b,
    const float* __restrict__ tW2_w, const float* __restrict__ tW2_b,
    const float* __restrict__ tV_w,
    const float* __restrict__ ih_w,  const float* __restrict__ ih_b,
    const float* __restrict__ hh_b,
    const float* __restrict__ dW1_w, const float* __restrict__ dW1_b,
    const float* __restrict__ dW2_w, const float* __restrict__ dW2_b,
    const float* __restrict__ dV_w,
    const float* __restrict__ pred_w, const float* __restrict__ pred_b,
    float* __restrict__ out) {
    extern __shared__ char smem[];
    _Float16* W4    = (_Float16*)smem;        // [256][LD2] Wall_t, later ih_t
    _Float16* W1    = W4 + 256 * LD2;         // [64][LD2]  Wd_t, later tW2_t
    _Float16* fullH = W1 + 64 * LD2;          // [304][LD2] all h states (f16)
    _Float16* hA    = fullH + 304 * LD2;      // [16][LD2]  h as A operand
    _Float16* cA    = hA + 16 * LD2;          // [16][LD2]  c as A op, later day_vec
    float* cF     = (float*)(cA + 16 * LD2);  // [16][64]
    float* hF     = cF + 16 * 64;             // [16][64]
    float* gatesF = hF + 16 * 64;             // [16][256]
    float* a1F    = gatesF + 16 * 256;        // [16][64]  (cs1 / a1 / te reuse)
    float* hdF    = a1F + 16 * 64;            // [16][64]
    float* dvF    = hdF + 16 * 64;            // [16][64]
    float* scoreF = dvF + 16 * 64;            // [320]
    float* wgtF   = scoreF + 320;             // [16]

    const int s    = blockIdx.x;
    const int tid  = threadIdx.x;
    const int lane = tid & 31;
    const int w    = tid >> 5;
    const int hi   = lane >> 4, nlo = lane & 15;

    // ---- Phase 0: stage weights + zero state ------------------------------
#pragma unroll 4
    for (int it = 0; it < 64; ++it) {          // Wall_w [64][256] -> W4[n][k]
        int j = tid + it * 256;
        W4[(j & 255) * LD2 + (j >> 8)] = (_Float16)Wall_w[(size_t)s * 16384 + j];
    }
#pragma unroll 4
    for (int it = 0; it < 16; ++it) {          // Wd_w [64][64] -> W1[n][k]
        int j = tid + it * 256;
        W1[(j & 63) * LD2 + (j >> 6)] = (_Float16)Wd_w[(size_t)s * 4096 + j];
    }
    for (int i = tid; i < 16 * LD2; i += 256) { hA[i] = (_Float16)0.f; cA[i] = (_Float16)0.f; }
    for (int i = tid; i < 16 * 64;  i += 256) { cF[i] = 0.f; hF[i] = 0.f; }
    for (int i = tid; i < 4 * LD2;  i += 256) fullH[300 * LD2 + i] = (_Float16)0.f;
    __syncthreads();

    // ---- Phase 1: TimeLSTM scan over T=30 --------------------------------
    for (int t = 0; t < T_; ++t) {
        // gates = h @ Wall  (16 col tiles over 8 waves, K=64)
#pragma unroll
        for (int q = 0; q < 2; ++q) {
            int ct = w + q * 8;
            v8f acc = vzero8();
#pragma unroll
            for (int ks = 0; ks < 2; ++ks) {
                v16h a = frag_A(hA, LD2, 0, ks * 32, lane);
                v16h b = frag_B(W4, LD2, ct * 16, ks * 32, lane);
                acc = __builtin_amdgcn_wmma_f32_16x16x32_f16(
                    false, a, false, b, (short)0, acc, false, false);
            }
            int n = ct * 16 + nlo;
            float bias = Wall_b[(size_t)s * G4_ + n];
#pragma unroll
            for (int v = 0; v < 8; ++v) {
                int m = v + hi * 8;
                if (m < L_) {
                    float ug = (float)u[((size_t)s * M_ + m * T_ + t) * G4_ + n];
                    gatesF[m * G4_ + n] = sigmoidf_(acc[v] + bias + ug);
                }
            }
        }
        // c_s1 = tanh(c @ Wd + Wd_b)  (4 col tiles on waves 0..3)
        if (w < 4) {
            int ct = w;
            v8f acc = vzero8();
#pragma unroll
            for (int ks = 0; ks < 2; ++ks) {
                v16h a = frag_A(cA, LD2, 0, ks * 32, lane);
                v16h b = frag_B(W1, LD2, ct * 16, ks * 32, lane);
                acc = __builtin_amdgcn_wmma_f32_16x16x32_f16(
                    false, a, false, b, (short)0, acc, false, false);
            }
            int n = ct * 16 + nlo;
            float bias = Wd_b[(size_t)s * H_ + n];
#pragma unroll
            for (int v = 0; v < 8; ++v) {
                int m = v + hi * 8;
                if (m < L_) a1F[m * H_ + n] = tanhf(acc[v] + bias);
            }
        }
        __syncthreads();
        // elementwise state update
        for (int i = tid; i < L_ * H_; i += 256) {
            int l = i >> 6, h = i & 63;
            float c  = cF[l * H_ + h];
            float c1 = a1F[l * H_ + h];
            float ts = time_in[(size_t)s * M_ + l * T_ + t];
            float cadj = c - c1 + c1 * ts;
            float fg = gatesF[l * G4_ + h];
            float ig = gatesF[l * G4_ + 64 + h];
            float og = gatesF[l * G4_ + 128 + h];
            float gg = gatesF[l * G4_ + 192 + h];
            float cn = fg * cadj + ig * gg;
            float hn = og * tanhf(cn);
            cF[l * H_ + h] = cn;  hF[l * H_ + h] = hn;
            cA[l * LD2 + h] = (_Float16)cn;
            hA[l * LD2 + h] = (_Float16)hn;
            fullH[(l * T_ + t) * LD2 + h] = (_Float16)hn;
        }
        __syncthreads();
    }

    // ---- Phase 2: text attention ------------------------------------------
#pragma unroll 4
    for (int it = 0; it < 16; ++it) {          // tW2 -> W1 transposed
        int j = tid + it * 256;
        W1[(j & 63) * LD2 + (j >> 6)] = (_Float16)tW2_w[(size_t)s * 4096 + j];
    }
    for (int i = tid; i < L_ * H_; i += 256) { // a1 = h_last @ tW1 + b (VALU)
        int l = i >> 6, k = i & 63;
        float acc = tW1_b[(size_t)s * H_ + k];
        for (int h = 0; h < H_; ++h)
            acc += hF[l * H_ + h] * tW1_w[(size_t)s * 4096 + h * H_ + k];
        a1F[l * H_ + k] = acc;
    }
    for (int i = tid; i < 320; i += 256) scoreF[i] = 0.f;
    __syncthreads();

    // score[m] += tanh(a1 + full@tW2 + tW2_b) * tV   (19x4 tiles over 8 waves)
    for (int tile = w; tile < 76; tile += 8) {
        int rt = tile >> 2, ct = tile & 3;
        v8f acc = vzero8();
#pragma unroll
        for (int ks = 0; ks < 2; ++ks) {
            v16h a = frag_A(fullH, LD2, rt * 16, ks * 32, lane);
            v16h b = frag_B(W1,    LD2, ct * 16, ks * 32, lane);
            acc = __builtin_amdgcn_wmma_f32_16x16x32_f16(
                false, a, false, b, (short)0, acc, false, false);
        }
        int n = ct * 16 + nlo;
        float w2b = tW2_b[(size_t)s * H_ + n];
        float tv  = tV_w[(size_t)s * H_ + n];
#pragma unroll
        for (int v = 0; v < 8; ++v) {
            int m = rt * 16 + v + hi * 8;
            if (m < M_) {
                int l = m / T_;
                float sc = tanhf(a1F[l * H_ + n] + acc[v] + w2b) * tv;
                atomicAdd(&scoreF[m], sc);
            }
        }
    }
    __syncthreads();
    if (tid < L_) {                            // softmax over T per day
        int l = tid;
        float mx = -1e30f;
        for (int t = 0; t < T_; ++t) mx = fmaxf(mx, scoreF[l * T_ + t]);
        float sm = 0.f;
        for (int t = 0; t < T_; ++t) {
            float e = __expf(scoreF[l * T_ + t] - mx);
            scoreF[l * T_ + t] = e; sm += e;
        }
        float inv = 1.f / sm;
        for (int t = 0; t < T_; ++t) scoreF[l * T_ + t] *= inv;
    }
    __syncthreads();
    for (int i = tid; i < L_ * H_; i += 256) { // day_vec = sum_t attn*full
        int l = i >> 6, h = i & 63;
        float acc = 0.f;
        for (int t = 0; t < T_; ++t)
            acc += scoreF[l * T_ + t] * (float)fullH[(l * T_ + t) * LD2 + h];
        dvF[l * H_ + h] = acc;
        cA[l * LD2 + h] = (_Float16)acc;       // cA rows 10..15 still zero
    }
    __syncthreads();

    // ---- Phase 3: day LSTM (single step from zero state) ------------------
#pragma unroll 4
    for (int it = 0; it < 64; ++it) {          // ih_w -> W4 transposed
        int j = tid + it * 256;
        W4[(j & 255) * LD2 + (j >> 8)] = (_Float16)ih_w[(size_t)s * 16384 + j];
    }
    __syncthreads();
#pragma unroll
    for (int q = 0; q < 2; ++q) {
        int ct = w + q * 8;
        v8f acc = vzero8();
#pragma unroll
        for (int ks = 0; ks < 2; ++ks) {
            v16h a = frag_A(cA, LD2, 0, ks * 32, lane);
            v16h b = frag_B(W4, LD2, ct * 16, ks * 32, lane);
            acc = __builtin_amdgcn_wmma_f32_16x16x32_f16(
                false, a, false, b, (short)0, acc, false, false);
        }
        int n = ct * 16 + nlo;
        float bias = ih_b[(size_t)s * G4_ + n] + hh_b[(size_t)s * G4_ + n];
#pragma unroll
        for (int v = 0; v < 8; ++v) {
            int m = v + hi * 8;
            if (m < L_) gatesF[m * G4_ + n] = acc[v] + bias;
        }
    }
    __syncthreads();
    for (int i = tid; i < L_ * H_; i += 256) { // torch gate order i,f,g,o
        int l = i >> 6, h = i & 63;
        float ig = sigmoidf_(gatesF[l * G4_ + h]);
        float gg = tanhf(gatesF[l * G4_ + 128 + h]);
        float og = sigmoidf_(gatesF[l * G4_ + 192 + h]);
        hdF[l * H_ + h] = og * tanhf(ig * gg);
    }
    __syncthreads();

    // ---- Phase 4: day attention + head ------------------------------------
    for (int i = tid; i < L_ * H_; i += 256) { // te = tanh(b1+b2)
        int l = i >> 6, k = i & 63;
        float acc = dW1_b[(size_t)s * H_ + k] + dW2_b[(size_t)s * H_ + k];
        for (int h = 0; h < H_; ++h)
            acc += hdF[l * H_ + h] * (dW1_w[(size_t)s * 4096 + h * H_ + k] +
                                      dW2_w[(size_t)s * 4096 + h * H_ + k]);
        a1F[l * H_ + k] = tanhf(acc);
    }
    __syncthreads();
    if (tid < L_) {
        int l = tid; float acc = 0.f;
        for (int k = 0; k < H_; ++k) acc += a1F[l * H_ + k] * dV_w[(size_t)s * H_ + k];
        wgtF[l] = acc;
    }
    __syncthreads();
    if (tid == 0) {                            // softmax over L
        float mx = -1e30f;
        for (int l = 0; l < L_; ++l) mx = fmaxf(mx, wgtF[l]);
        float sm = 0.f;
        for (int l = 0; l < L_; ++l) { float e = __expf(wgtF[l] - mx); wgtF[l] = e; sm += e; }
        float inv = 1.f / sm;
        for (int l = 0; l < L_; ++l) wgtF[l] *= inv;
    }
    __syncthreads();
    if (tid < H_) {
        int h = tid; float acc = 0.f;
        for (int l = 0; l < L_; ++l) acc += wgtF[l] * hdF[l * H_ + h];
        dvF[h] = acc * pred_w[h];
    }
    __syncthreads();
    if (tid == 0) {
        float acc = pred_b[0];
        for (int h = 0; h < H_; ++h) acc += dvF[h];
        out[s] = acc > 0.f ? acc : 0.01f * acc;
    }
}

// ===========================================================================
extern "C" void kernel_launch(void* const* d_in, const int* in_sizes, int n_in,
                              void* d_out, int out_size, void* d_ws, size_t ws_size,
                              hipStream_t stream) {
    (void)in_sizes; (void)n_in; (void)out_size; (void)ws_size;
    const float* text   = (const float*)d_in[0];
    const float* time_i = (const float*)d_in[1];
    const float* Wd_w   = (const float*)d_in[2];
    const float* Wd_b   = (const float*)d_in[3];
    const float* Wall_w = (const float*)d_in[4];
    const float* Wall_b = (const float*)d_in[5];
    const float* Uall_w = (const float*)d_in[6];
    const float* Uall_b = (const float*)d_in[7];
    const float* tW1_w  = (const float*)d_in[8];
    const float* tW1_b  = (const float*)d_in[9];
    const float* tW2_w  = (const float*)d_in[10];
    const float* tW2_b  = (const float*)d_in[11];
    const float* tV_w   = (const float*)d_in[12];
    const float* ih_w   = (const float*)d_in[14];
    const float* ih_b   = (const float*)d_in[15];
    const float* hh_b   = (const float*)d_in[16];
    const float* dW1_w  = (const float*)d_in[17];
    const float* dW1_b  = (const float*)d_in[18];
    const float* dW2_w  = (const float*)d_in[19];
    const float* dW2_b  = (const float*)d_in[20];
    const float* dV_w   = (const float*)d_in[21];
    const float* pred_w = (const float*)d_in[23];
    const float* pred_b = (const float*)d_in[24];
    float* out = (float*)d_out;

    // workspace carve (f16): Xh [S][300][768], Wt [S][256][768], Uh [S][300][256]
    _Float16* Xh = (_Float16*)d_ws;
    _Float16* Wt = Xh + (size_t)S_ * M_ * E_;
    _Float16* Uh = Wt + (size_t)S_ * G4_ * E_;

    // S*300*768 = 20,275,200 elems = 2,534,400 v8h = 9900 blocks of 256
    conv_x<<<9900, 256, 0, stream>>>(text, Xh);
    conv_w<<<dim3(48, S_), 256, 0, stream>>>(Uall_w, Wt);

    dim3 g1(5, S_);
    fast_ugemm<<<g1, 256, 0, stream>>>(Xh, Wt, Uall_b, Uh);

    size_t smem2 = (size_t)(256 + 64 + 304 + 16 + 16) * LD2 * sizeof(_Float16)
                 + (size_t)(16 * 64 * 5 + 16 * 256 + 320 + 16) * sizeof(float);
    fast_stock<<<S_, 256, smem2, stream>>>(
        Uh, time_i, Wd_w, Wd_b, Wall_w, Wall_b,
        tW1_w, tW1_b, tW2_w, tW2_b, tV_w,
        ih_w, ih_b, hh_b, dW1_w, dW1_b, dW2_w, dW2_b, dV_w,
        pred_w, pred_b, out);
}
